// LSTM_18708877541498
// MI455X (gfx1250) — matrix operand order repaired and checked by reading, not compiled, not run on hardware
//
#include <hip/hip_runtime.h>
#include <hip/hip_bf16.h>

typedef __attribute__((ext_vector_type(16))) __bf16 v16bf;
typedef __attribute__((ext_vector_type(8)))  __bf16 v8bf;
typedef __attribute__((ext_vector_type(8)))  float  v8f;

#define B_DIM 4096
#define D_IN  4096
#define E_DIM 2048
#define H_DIM 2048
#define V_DIM 32000

// ---------------------------------------------------------------------------
// fp32 -> bf16 convert, 8 elements/thread (cols and dstStride multiples of 8)
// ---------------------------------------------------------------------------
__global__ void cvt_kernel(const float* __restrict__ src, __bf16* __restrict__ dst,
                           long rows, long cols, long dstStride) {
    long i = ((long)blockIdx.x * blockDim.x + threadIdx.x) * 8;
    if (i < rows * cols) {
        long r = i / cols, c = i - r * cols;
        float4 f0 = *(const float4*)&src[i];
        float4 f1 = *(const float4*)&src[i + 4];
        v8bf o;
        o[0] = (__bf16)f0.x; o[1] = (__bf16)f0.y; o[2] = (__bf16)f0.z; o[3] = (__bf16)f0.w;
        o[4] = (__bf16)f1.x; o[5] = (__bf16)f1.y; o[6] = (__bf16)f1.z; o[7] = (__bf16)f1.w;
        *(v8bf*)&dst[r * dstStride + c] = o;
    }
}

// ---------------------------------------------------------------------------
// fp32 -> bf16 transpose-convert: dst[c*dstStride + r] = src[r*C + c]
// ---------------------------------------------------------------------------
__global__ void tcvt_kernel(const float* __restrict__ src, __bf16* __restrict__ dst,
                            long R, long C, long dstStride) {
    __shared__ float tile[32][33];
    long c0 = (long)blockIdx.x * 32, r0 = (long)blockIdx.y * 32;
    int tx = threadIdx.x, ty = threadIdx.y;
#pragma unroll
    for (int i = 0; i < 32; i += 8)
        tile[ty + i][tx] = src[(r0 + ty + i) * C + c0 + tx];
    __syncthreads();
#pragma unroll
    for (int i = 0; i < 32; i += 8)
        dst[(c0 + ty + i) * dstStride + r0 + tx] = (__bf16)tile[tx][ty + i];
}

// ---------------------------------------------------------------------------
// combined gate bias
// ---------------------------------------------------------------------------
__global__ void bias_kernel(const float* bf_, const float* bhf, const float* bi_, const float* bhi,
                            const float* bc_, const float* bhc, const float* bo_, const float* bho,
                            float* __restrict__ out, int H) {
    int i = blockIdx.x * blockDim.x + threadIdx.x;
    if (i < H) {
        out[i]         = bf_[i] + bhf[i];
        out[H + i]     = bi_[i] + bhi[i];
        out[2 * H + i] = bc_[i] + bhc[i];
        out[3 * H + i] = bo_[i] + bho[i];
    }
}

// ---------------------------------------------------------------------------
// bf16 GEMM: C[M,N] = A[M,K] @ Bt[N,K]^T + bias[N]
// Block tile 128(M) x 256(N) x 32(K), 256 threads = 8 wave32.
// Wave (waveM 0..1, waveN 0..3) owns 64x64 = 4x4 WMMA tiles (16 wmma/k-step).
// Register-staged double-buffered LDS: one barrier per k-step; next tile's
// global loads are issued before the WMMA block so HBM latency overlaps MFU.
// BF16OUT selects the (branchless) epilogue store type.
// ---------------------------------------------------------------------------
#define BK     32
#define LDS_S  40   // BK + 8 pad; 80B row pitch keeps 16B alignment

template <bool BF16OUT>
__global__ __launch_bounds__(256)
void gemm_bf16_kernel(const __bf16* __restrict__ A, const __bf16* __restrict__ Bt,
                      const float* __restrict__ bias, void* __restrict__ outP,
                      int K, long ldOut) {
    __shared__ __bf16 sA[2][128 * LDS_S];
    __shared__ __bf16 sB[2][256 * LDS_S];

    const int tid  = threadIdx.x;
    const int lane = tid & 31;
    const int wave = tid >> 5;
    const int waveM = wave & 1;   // 0..1 -> M offset 64*waveM
    const int waveN = wave >> 1;  // 0..3 -> N offset 64*waveN
    const int l16   = lane & 15;
    const int hsel  = lane >> 4;

    const long mBlock = (long)blockIdx.y * 128;
    const long nBlock = (long)blockIdx.x * 256;

    // cooperative-load indexing: 16B chunks, K-contiguous
    const int ldRow = tid >> 2;           // 0..63
    const int ldCol = (tid & 3) * 8;

    v8f acc[4][4];
#pragma unroll
    for (int mt = 0; mt < 4; ++mt)
#pragma unroll
        for (int nt = 0; nt < 4; ++nt)
            acc[mt][nt] = (v8f)0.0f;

    v8bf aStage[2], bStage[4];

    auto loadStage = [&](int k0) {
#pragma unroll
        for (int p = 0; p < 2; ++p)
            aStage[p] = *(const v8bf*)&A[(mBlock + ldRow + p * 64) * (long)K + k0 + ldCol];
#pragma unroll
        for (int p = 0; p < 4; ++p)
            bStage[p] = *(const v8bf*)&Bt[(nBlock + ldRow + p * 64) * (long)K + k0 + ldCol];
    };
    auto storeStage = [&](int buf) {
#pragma unroll
        for (int p = 0; p < 2; ++p)
            *(v8bf*)&sA[buf][(ldRow + p * 64) * LDS_S + ldCol] = aStage[p];
#pragma unroll
        for (int p = 0; p < 4; ++p)
            *(v8bf*)&sB[buf][(ldRow + p * 64) * LDS_S + ldCol] = bStage[p];
    };

    loadStage(0);
    storeStage(0);
    __syncthreads();

    const int nk = K / BK;
    for (int kt = 0; kt < nk; ++kt) {
        const int cur = kt & 1;
        const bool hasNext = (kt + 1 < nk);
        if (hasNext) loadStage((kt + 1) * BK);   // global loads in flight over WMMAs

        // A fragments: lane row = l16, K halves {hsel*8..+7} and {+16..}
        v16bf af[4];
#pragma unroll
        for (int mt = 0; mt < 4; ++mt) {
            int r  = waveM * 64 + mt * 16 + l16;
            int kb = hsel * 8;
            v8bf lo = *(const v8bf*)&sA[cur][r * LDS_S + kb];
            v8bf hi = *(const v8bf*)&sA[cur][r * LDS_S + kb + 16];
            af[mt] = __builtin_shufflevector(lo, hi, 0,1,2,3,4,5,6,7,8,9,10,11,12,13,14,15);
        }
        // B fragments: lane col = l16, K halves {hsel*8..+7} and {+16..}
        v16bf bfv[4];
#pragma unroll
        for (int nt = 0; nt < 4; ++nt) {
            int r  = waveN * 64 + nt * 16 + l16;
            int kb = hsel * 8;
            v8bf lo = *(const v8bf*)&sB[cur][r * LDS_S + kb];
            v8bf hi = *(const v8bf*)&sB[cur][r * LDS_S + kb + 16];
            bfv[nt] = __builtin_shufflevector(lo, hi, 0,1,2,3,4,5,6,7,8,9,10,11,12,13,14,15);
        }
#pragma unroll
        for (int mt = 0; mt < 4; ++mt)
#pragma unroll
            for (int nt = 0; nt < 4; ++nt)
                acc[mt][nt] = __builtin_amdgcn_wmma_f32_16x16x32_bf16(
                    false, af[mt], false, bfv[nt], (short)0, acc[mt][nt], false, false);

        if (hasNext) storeStage(cur ^ 1);
        __syncthreads();
    }

    // ---- branchless epilogue; C/D layout: lane col = l16, VGPR r -> row r+8*hsel
#pragma unroll
    for (int nt = 0; nt < 4; ++nt) {
        long n = nBlock + waveN * 64 + nt * 16 + l16;
        float bv = bias[n];
        long mBase = mBlock + waveM * 64 + 8 * hsel;
#pragma unroll
        for (int mt = 0; mt < 4; ++mt) {
#pragma unroll
            for (int r = 0; r < 8; ++r) {
                long m = mBase + mt * 16 + r;
                float v = acc[mt][nt][r] + bv;
                if (BF16OUT) ((__bf16*)outP)[m * ldOut + n] = (__bf16)v;
                else         ((float*)outP)[m * ldOut + n] = v;
            }
        }
    }
}

// ---------------------------------------------------------------------------
// gate nonlinearity + cell/hidden update
// ---------------------------------------------------------------------------
__global__ void gates_kernel(const float* __restrict__ pre, const float* __restrict__ cell,
                             float* __restrict__ outH, float* __restrict__ outC,
                             __bf16* __restrict__ outHb, long Bn, long H) {
    long i = (long)blockIdx.x * blockDim.x + threadIdx.x;
    if (i < Bn * H) {
        long b = i / H, h = i - b * H;
        const float* row = pre + b * 4 * H;
        float f  = 1.0f / (1.0f + expf(-row[h]));
        float in = 1.0f / (1.0f + expf(-row[H + h]));
        float g  = tanhf(row[2 * H + h]);
        float o  = 1.0f / (1.0f + expf(-row[3 * H + h]));
        float c  = f * cell[i] + in * g;
        float hh = o * tanhf(c);
        outC[i]  = c;
        outH[i]  = hh;
        outHb[i] = (__bf16)hh;
    }
}

// ---------------------------------------------------------------------------
// in-place log_softmax over rows of length V; one 1024-thread block per row
// ---------------------------------------------------------------------------
__global__ __launch_bounds__(1024)
void lsm_kernel(float* __restrict__ logits, int V) {
    __shared__ float red[1024];
    float* x = logits + (long)blockIdx.x * V;
    int t = threadIdx.x;

    float mx = -INFINITY;
    for (int j = t; j < V; j += 1024) mx = fmaxf(mx, x[j]);
    red[t] = mx; __syncthreads();
    for (int s = 512; s > 0; s >>= 1) { if (t < s) red[t] = fmaxf(red[t], red[t + s]); __syncthreads(); }
    mx = red[0]; __syncthreads();

    float sum = 0.0f;
    for (int j = t; j < V; j += 1024) sum += expf(x[j] - mx);
    red[t] = sum; __syncthreads();
    for (int s = 512; s > 0; s >>= 1) { if (t < s) red[t] += red[t + s]; __syncthreads(); }
    float lse = mx + logf(red[0]);

    for (int j = t; j < V; j += 1024) x[j] = x[j] - lse;
}

// ---------------------------------------------------------------------------
extern "C" void kernel_launch(void* const* d_in, const int* in_sizes, int n_in,
                              void* d_out, int out_size, void* d_ws, size_t ws_size,
                              hipStream_t stream) {
    (void)in_sizes; (void)n_in; (void)out_size; (void)ws_size;
    const float* input  = (const float*)d_in[0];
    const float* hidden = (const float*)d_in[1];
    const float* cell   = (const float*)d_in[2];
    const float* We  = (const float*)d_in[3];  const float* be   = (const float*)d_in[4];
    const float* Wf  = (const float*)d_in[5];  const float* bf_  = (const float*)d_in[6];
    const float* Wi  = (const float*)d_in[7];  const float* bi_  = (const float*)d_in[8];
    const float* Wc  = (const float*)d_in[9];  const float* bc_  = (const float*)d_in[10];
    const float* Wo  = (const float*)d_in[11]; const float* bo_  = (const float*)d_in[12];
    const float* Uf  = (const float*)d_in[13]; const float* bhf  = (const float*)d_in[14];
    const float* Ui  = (const float*)d_in[15]; const float* bhi  = (const float*)d_in[16];
    const float* Uc  = (const float*)d_in[17]; const float* bhc  = (const float*)d_in[18];
    const float* Uo  = (const float*)d_in[19]; const float* bho  = (const float*)d_in[20];
    const float* Wend = (const float*)d_in[21]; const float* bend = (const float*)d_in[22];

    const long Bn = B_DIM, Din = D_IN, E = E_DIM, H = H_DIM, V = V_DIM;
    const long EH = E + H;

    char* ws = (char*)d_ws;
    auto carve = [&](size_t bytes) { char* p = ws; ws += (bytes + 255) & ~(size_t)255; return p; };
    __bf16* inB   = (__bf16*)carve((size_t)Bn * Din * 2);
    __bf16* WeT   = (__bf16*)carve((size_t)E * Din * 2);
    __bf16* WgT   = (__bf16*)carve((size_t)4 * H * EH * 2);
    __bf16* WendT = (__bf16*)carve((size_t)V * H * 2);
    __bf16* Abig  = (__bf16*)carve((size_t)Bn * EH * 2);
    float*  preact= (float*) carve((size_t)Bn * 4 * H * 4);
    float*  biasg = (float*) carve((size_t)4 * H * 4);
    __bf16* hB    = (__bf16*)carve((size_t)Bn * H * 2);

    float* outLogits = (float*)d_out;
    float* outHidden = outLogits + (size_t)Bn * V;
    float* outCell   = outHidden + (size_t)Bn * H;

    dim3 blk256(256);

    cvt_kernel<<<dim3((unsigned)((Bn * Din / 8 + 255) / 256)), blk256, 0, stream>>>(
        input, inB, Bn, Din, Din);
    cvt_kernel<<<dim3((unsigned)((Bn * H / 8 + 255) / 256)), blk256, 0, stream>>>(
        hidden, Abig + E, Bn, H, EH);

    dim3 tblk(32, 8);
    tcvt_kernel<<<dim3((unsigned)(E / 32), (unsigned)(Din / 32)), tblk, 0, stream>>>(
        We, WeT, Din, E, Din);
    const float* Wg[4] = {Wf, Wi, Wc, Wo};
    const float* Ug[4] = {Uf, Ui, Uc, Uo};
    for (int gi = 0; gi < 4; ++gi) {
        tcvt_kernel<<<dim3((unsigned)(H / 32), (unsigned)(Din / 32)), tblk, 0, stream>>>(
            Wg[gi], WgT + (size_t)gi * H * EH, E, H, EH);
        tcvt_kernel<<<dim3((unsigned)(H / 32), (unsigned)(H / 32)), tblk, 0, stream>>>(
            Ug[gi], WgT + (size_t)gi * H * EH + E, H, H, EH);
    }
    tcvt_kernel<<<dim3((unsigned)(V / 32), (unsigned)(H / 32)), tblk, 0, stream>>>(
        Wend, WendT, H, V, H);

    bias_kernel<<<dim3((unsigned)((H + 255) / 256)), blk256, 0, stream>>>(
        bf_, bhf, bi_, bhi, bc_, bhc, bo_, bho, biasg, (int)H);

    // emb = input @ We + be -> bf16 into left half of Abig (stride EH)
    gemm_bf16_kernel<true><<<dim3((unsigned)(E / 256), (unsigned)(Bn / 128)), blk256, 0, stream>>>(
        inB, WeT, be, Abig, (int)Din, EH);

    // gate pre-activations: [B, 4H] = Abig @ WgT^T + biasg (fp32)
    gemm_bf16_kernel<false><<<dim3((unsigned)(4 * H / 256), (unsigned)(Bn / 128)), blk256, 0, stream>>>(
        Abig, WgT, biasg, preact, (int)EH, 4 * H);

    gates_kernel<<<dim3((unsigned)((Bn * H + 255) / 256)), blk256, 0, stream>>>(
        preact, cell, outHidden, outCell, hB, Bn, H);

    // logits = new_hidden @ Wend + bend -> fp32 into d_out
    gemm_bf16_kernel<false><<<dim3((unsigned)(V / 256), (unsigned)(Bn / 128)), blk256, 0, stream>>>(
        hB, WendT, bend, outLogits, (int)H, V);

    lsm_kernel<<<dim3((unsigned)Bn), dim3(1024), 0, stream>>>(outLogits, (int)V);
}
